// DualEncoderMatchingAE_39204461478076
// MI455X (gfx1250) — matrix operand order repaired and checked by reading, not compiled, not run on hardware
//
#include <hip/hip_runtime.h>
#include <hip/hip_bf16.h>

#define B_N 2048
#define TOPK 10

typedef __attribute__((ext_vector_type(16))) _Float16 v16h;
typedef __attribute__((ext_vector_type(8)))  _Float16 v8h;
typedef __attribute__((ext_vector_type(8)))  float    v8f;

#if defined(__gfx1250__) && __has_builtin(__builtin_amdgcn_global_load_async_to_lds_b128)
#define HAVE_ASYNC_LDS 1
// Builtin signature (from hipcc diagnostic): param 1 is
// 'int __attribute__((vector_size(16))) __device__ *'  i.e. global int4*,
// param 2 is the LDS-side pointer.  Build exactly those types.
typedef int v4i_ __attribute__((vector_size(16)));
typedef __attribute__((address_space(1))) v4i_ as1_v4i;   // global
typedef __attribute__((address_space(3))) v4i_ as3_v4i;   // LDS
#endif

// ---------------------------------------------------------------------------
// WMMA fragment helpers (wave32; A/B use contiguous k, not the ISA table order)
// A (16x32 f16): lane L -> m = L&15, grp = L>>4;
//                K = grp*16 + e, e=0..15  => two contiguous 8-half chunks.
// B (32x16 f16, stored as Bt[N][K] row-major): lane L -> n = L&15,
//                K = (L>>4)*16 + e, e=0..15 => one contiguous 16-half chunk.
// C/D (16x16 f32): lane L, vgpr v -> n = L&15, m = v + (L>>4)*8.
// ---------------------------------------------------------------------------
__device__ __forceinline__ v16h combine16(v8h lo, v8h hi) {
  v16h r;
#pragma unroll
  for (int i = 0; i < 8; ++i) { r[i] = lo[i]; r[i + 8] = hi[i]; }
  return r;
}

__device__ __forceinline__ v16h load_a_frag(const _Float16* A, int ldk,
                                            int m_base, int k0, int lane) {
  int m = m_base + (lane & 15);
  int grp = lane >> 4;
  const _Float16* p = A + (size_t)m * ldk + k0 + grp * 16;
  v8h lo = *(const v8h*)p;
  v8h hi = *(const v8h*)(p + 8);
  return combine16(lo, hi);
}

__device__ __forceinline__ v16h load_bt_frag(const _Float16* Bt, int ldk,
                                             int n_base, int k0, int lane) {
  int n = n_base + (lane & 15);
  int kb = k0 + (lane >> 4) * 16;
  return *(const v16h*)(Bt + (size_t)n * ldk + kb);
}

__device__ __forceinline__ v8f wmma_f16(v16h a, v16h b, v8f c) {
  return __builtin_amdgcn_wmma_f32_16x16x32_f16(false, a, false, b,
                                                (short)0, c, false, false);
}

// ---------------------------------------------------------------------------
// Small conversion / packing kernels
// ---------------------------------------------------------------------------
__global__ void k_f32_to_f16(const float* __restrict__ in,
                             _Float16* __restrict__ out, int n) {
  int i = blockIdx.x * blockDim.x + threadIdx.x;
  if (i < n) out[i] = (_Float16)in[i];
}

// (64,32,3,3) OIHW f32 -> f16 [tap][co][ci]  (Bt-friendly: contiguous ci)
__global__ void k_pack_w2(const float* __restrict__ w,
                          _Float16* __restrict__ out) {
  int t = blockIdx.x * blockDim.x + threadIdx.x;
  if (t >= 9 * 64 * 32) return;
  int tap = t / (64 * 32);
  int rem = t % (64 * 32);
  int co = rem / 32, ci = rem % 32;
  out[t] = (_Float16)w[(co * 32 + ci) * 9 + tap];
}

// ---------------------------------------------------------------------------
// conv1: 1->32, 3x3 SAME over an (optionally cropped) window; channel-last f16
// ---------------------------------------------------------------------------
__global__ void k_conv1(const float* __restrict__ x, int y0, int x0,
                        int Ho, int Wo, const float* __restrict__ w,
                        const float* __restrict__ bias,
                        _Float16* __restrict__ out) {
  int t = blockIdx.x * blockDim.x + threadIdx.x;
  int total = B_N * Ho * Wo;
  if (t >= total) return;
  int b = t / (Ho * Wo);
  int p = t % (Ho * Wo);
  int y = p / Wo, xo = p % Wo;
  const int H = 28, W = 28;
  float acc[32];
#pragma unroll
  for (int c = 0; c < 32; ++c) acc[c] = bias[c];
  for (int dy = 0; dy < 3; ++dy) {
    int iy = y + dy - 1;
    if (iy < 0 || iy >= Ho) continue;
    for (int dx = 0; dx < 3; ++dx) {
      int ix = xo + dx - 1;
      if (ix < 0 || ix >= Wo) continue;
      float xv = x[((size_t)b * H + (y0 + iy)) * W + (x0 + ix)];
#pragma unroll
      for (int c = 0; c < 32; ++c) acc[c] += xv * w[c * 9 + dy * 3 + dx];
    }
  }
  _Float16* o = out + (size_t)t * 32;
#pragma unroll
  for (int c = 0; c < 32; ++c) {
    float v = acc[c];
    o[c] = (_Float16)(v > 0.f ? v : 0.f);
  }
}

// ---------------------------------------------------------------------------
// Fused conv2 (32->64, 3x3 SAME, implicit GEMM via WMMA) + relu + global mean
// pool + FC(64->64).  One workgroup (4 waves) per image.  Weights staged in
// LDS via the gfx1250 async global->LDS path when available.
// ---------------------------------------------------------------------------
__global__ void k_conv2_pool_fc(const _Float16* __restrict__ h1, int Ho, int Wo,
                                const _Float16* __restrict__ wpk,
                                const float* __restrict__ bias2,
                                const float* __restrict__ fcw,
                                const float* __restrict__ fcb,
                                float* __restrict__ z, int zoff) {
  __shared__ _Float16 wlds[9 * 64 * 32];   // [tap][co][ci] f16, 36 KB
  __shared__ float feat[64];
  int b = blockIdx.x;
  int tid = threadIdx.x;                   // 128 threads = 4 waves
#if HAVE_ASYNC_LDS
  {
    // 36864 B = 2304 x b128 transfers; ASYNCcnt-tracked, no VGPR return path
    const char* gsrc = (const char*)wpk;
    char* ldst = (char*)wlds;
    for (int i = tid; i < 2304; i += 128) {
      __builtin_amdgcn_global_load_async_to_lds_b128(
          (as1_v4i*)(gsrc + (size_t)i * 16),
          (as3_v4i*)(ldst + (size_t)i * 16),
          0, 0);
    }
#if __has_builtin(__builtin_amdgcn_s_wait_asynccnt)
    __builtin_amdgcn_s_wait_asynccnt(0);
#else
    asm volatile("s_wait_asynccnt 0x0" ::: "memory");
#endif
  }
#else
  for (int i = tid; i < 9 * 64 * 32; i += 128) wlds[i] = wpk[i];
#endif
  if (tid < 64) feat[tid] = 0.f;
  __syncthreads();

  int lane = tid & 31, wave = tid >> 5;
  int npix = Ho * Wo;
  int ntiles = (npix + 15) >> 4;
  int mrow = lane & 15, grp = lane >> 4;

  for (int t = wave; t < ntiles; t += 4) {
    int p0 = t * 16;
    v8f acc0 = {}, acc1 = {}, acc2 = {}, acc3 = {};
    int p = p0 + mrow;
    int py = p / Wo, px = p % Wo;
    bool pvalid = p < npix;
#pragma unroll
    for (int tap = 0; tap < 9; ++tap) {
      int dy = tap / 3 - 1, dx = tap % 3 - 1;
      int iy = py + dy, ix = px + dx;
      bool inb = pvalid && iy >= 0 && iy < Ho && ix >= 0 && ix < Wo;
      v8h lo = {}, hi = {};
      if (inb) {
        const _Float16* ap =
            h1 + ((size_t)b * npix + iy * Wo + ix) * 32 + grp * 16;
        lo = *(const v8h*)ap;
        hi = *(const v8h*)(ap + 8);
      }
      v16h a = combine16(lo, hi);
      const _Float16* wb = wlds + tap * 2048;
      int kb = grp * 16;
      v16h b0 = *(const v16h*)(wb + (0 * 16 + mrow) * 32 + kb);
      v16h b1 = *(const v16h*)(wb + (1 * 16 + mrow) * 32 + kb);
      v16h b2 = *(const v16h*)(wb + (2 * 16 + mrow) * 32 + kb);
      v16h b3 = *(const v16h*)(wb + (3 * 16 + mrow) * 32 + kb);
      acc0 = wmma_f16(a, b0, acc0);
      acc1 = wmma_f16(a, b1, acc1);
      acc2 = wmma_f16(a, b2, acc2);
      acc3 = wmma_f16(a, b3, acc3);
    }
    // bias + relu + masked pixel-sum, pooled into LDS feat[]
    int m0 = grp * 8;
#pragma unroll
    for (int nt = 0; nt < 4; ++nt) {
      v8f acc = nt == 0 ? acc0 : nt == 1 ? acc1 : nt == 2 ? acc2 : acc3;
      int n = nt * 16 + mrow;
      float bv = bias2[n];
      float s = 0.f;
#pragma unroll
      for (int v = 0; v < 8; ++v) {
        if (p0 + m0 + v < npix) {
          float val = acc[v] + bv;
          s += val > 0.f ? val : 0.f;
        }
      }
      atomicAdd(&feat[n], s);
    }
  }
  __syncthreads();
  if (tid < 64) feat[tid] *= (1.0f / (float)npix);
  __syncthreads();
  if (tid < 64) {
    float a = fcb[tid];
#pragma unroll 8
    for (int j = 0; j < 64; ++j) a += fcw[tid * 64 + j] * feat[j];
    z[(size_t)b * 128 + zoff + tid] = a;
  }
}

// ---------------------------------------------------------------------------
// Row-normalize (D=128) f32 -> f16
// ---------------------------------------------------------------------------
__global__ void k_rownorm(const float* __restrict__ in,
                          _Float16* __restrict__ out) {
  int r = blockIdx.x, t = threadIdx.x;  // blockDim = 128
  __shared__ float red[128];
  float v = in[(size_t)r * 128 + t];
  red[t] = v * v;
  __syncthreads();
  for (int s = 64; s > 0; s >>= 1) {
    if (t < s) red[t] += red[t + s];
    __syncthreads();
  }
  float nrm = sqrtf(red[0]);
  nrm = nrm > 1e-12f ? nrm : 1e-12f;
  out[(size_t)r * 128 + t] = (_Float16)(v / nrm);
}

// ---------------------------------------------------------------------------
// GEMM  C[M,N] = A[M,128] * Bt[N,128]^T (+bias[n])
// One wave per 16x64 tile: A fragment reused across 4 B fragments -> 16 WMMAs
// per wave and 4x less A traffic than 16x16 tiling.  N must be % 64.
// ---------------------------------------------------------------------------
__global__ void k_gemm_nt(const _Float16* __restrict__ A,
                          const _Float16* __restrict__ Bt,
                          const float* __restrict__ bias,
                          float* __restrict__ C, int M, int N) {
  int wave = blockIdx.x * (blockDim.x >> 5) + (threadIdx.x >> 5);
  int lane = threadIdx.x & 31;
  int tiles_n = N >> 6;
  int tm = wave / tiles_n, tn = wave % tiles_n;
  if (tm >= (M >> 4)) return;
  int nb = tn * 64;
  __builtin_prefetch(A + (size_t)(tm * 16) * 128, 0, 1);
  __builtin_prefetch(Bt + (size_t)nb * 128, 0, 1);
  v8f c0 = {}, c1 = {}, c2 = {}, c3 = {};
#pragma unroll
  for (int kk = 0; kk < 4; ++kk) {
    int k0 = kk * 32;
    v16h a = load_a_frag(A, 128, tm * 16, k0, lane);
    v16h b0 = load_bt_frag(Bt, 128, nb + 0, k0, lane);
    v16h b1 = load_bt_frag(Bt, 128, nb + 16, k0, lane);
    v16h b2 = load_bt_frag(Bt, 128, nb + 32, k0, lane);
    v16h b3 = load_bt_frag(Bt, 128, nb + 48, k0, lane);
    c0 = wmma_f16(a, b0, c0);
    c1 = wmma_f16(a, b1, c1);
    c2 = wmma_f16(a, b2, c2);
    c3 = wmma_f16(a, b3, c3);
  }
  int m0 = tm * 16 + (lane >> 4) * 8;
  int nl = lane & 15;
#pragma unroll
  for (int nt = 0; nt < 4; ++nt) {
    v8f acc = nt == 0 ? c0 : nt == 1 ? c1 : nt == 2 ? c2 : c3;
    int n = nb + nt * 16 + nl;
    float bv = bias ? bias[n] : 0.f;
#pragma unroll
    for (int v = 0; v < 8; ++v) C[(size_t)(m0 + v) * N + n] = acc[v] + bv;
  }
}

// ---------------------------------------------------------------------------
// Per-row top-10 + softmax + weighted memory blend. One wave per row.
// ---------------------------------------------------------------------------
__global__ void k_topk_match(const float* __restrict__ sim,
                             const float* __restrict__ mem,
                             float* __restrict__ zm,
                             _Float16* __restrict__ zm16) {
  int b = blockIdx.x;
  int lane = threadIdx.x;  // blockDim = 32
  __shared__ float vbuf[32 * TOPK];
  __shared__ int ibuf[32 * TOPK];
  __shared__ float swt[TOPK];
  __shared__ int sid[TOPK];
  float tv[TOPK];
  int ti[TOPK];
#pragma unroll
  for (int j = 0; j < TOPK; ++j) { tv[j] = -3.4e38f; ti[j] = 0; }
  const float* row = sim + (size_t)b * 8192;
  for (int c = lane; c < 8192; c += 32) {
    float cv = row[c];
    int cidx = c;
#pragma unroll
    for (int j = 0; j < TOPK; ++j) {
      if (cv > tv[j]) {
        float tf = tv[j]; int tii = ti[j];
        tv[j] = cv; ti[j] = cidx;
        cv = tf; cidx = tii;
      }
    }
  }
#pragma unroll
  for (int j = 0; j < TOPK; ++j) {
    vbuf[lane * TOPK + j] = tv[j];
    ibuf[lane * TOPK + j] = ti[j];
  }
  __syncthreads();
  if (lane == 0) {
    float gv[TOPK]; int gi[TOPK];
#pragma unroll
    for (int j = 0; j < TOPK; ++j) { gv[j] = -3.4e38f; gi[j] = 0; }
    for (int e = 0; e < 32 * TOPK; ++e) {
      float cv = vbuf[e]; int cidx = ibuf[e];
#pragma unroll
      for (int j = 0; j < TOPK; ++j) {
        if (cv > gv[j]) {
          float tf = gv[j]; int tii = gi[j];
          gv[j] = cv; gi[j] = cidx;
          cv = tf; cidx = tii;
        }
      }
    }
    float mx = gv[0], s = 0.f, ex[TOPK];
#pragma unroll
    for (int j = 0; j < TOPK; ++j) { ex[j] = __expf(gv[j] - mx); s += ex[j]; }
    float inv = 1.f / s;
#pragma unroll
    for (int j = 0; j < TOPK; ++j) { swt[j] = ex[j] * inv; sid[j] = gi[j]; }
  }
  __syncthreads();
  for (int d = lane; d < 128; d += 32) {
    float a = 0.f;
#pragma unroll
    for (int j = 0; j < TOPK; ++j) a += swt[j] * mem[(size_t)sid[j] * 128 + d];
    zm[(size_t)b * 128 + d] = a;
    zm16[(size_t)b * 128 + d] = (_Float16)a;
  }
}

// ---------------------------------------------------------------------------
// conv_transpose 4x4 stride2 SAME (pad=1) + relu.  Plane-major layouts.
// in: (B,CI,IH,IW), w HWIO: (4,4,CI,CO), out: (B,CO,2IH,2IW)
// ---------------------------------------------------------------------------
__global__ void k_deconv(const float* __restrict__ in,
                         const float* __restrict__ w,
                         const float* __restrict__ bias,
                         float* __restrict__ out, int CI, int CO, int IH,
                         int IW) {
  int OH = IH * 2, OW = IW * 2;
  long long t = (long long)blockIdx.x * blockDim.x + threadIdx.x;
  long long total = (long long)B_N * CO * OH * OW;
  if (t >= total) return;
  int b = (int)(t / ((long long)CO * OH * OW));
  int rem = (int)(t % ((long long)CO * OH * OW));
  int co = rem / (OH * OW);
  int p = rem % (OH * OW);
  int oy = p / OW, ox = p % OW;
  float acc = bias[co];
  for (int ky = 0; ky < 4; ++ky) {
    int ty = oy + 1 - ky;
    if (ty & 1) continue;
    int iy = ty >> 1;
    if (iy < 0 || iy >= IH) continue;
    for (int kx = 0; kx < 4; ++kx) {
      int tx = ox + 1 - kx;
      if (tx & 1) continue;
      int ix = tx >> 1;
      if (ix < 0 || ix >= IW) continue;
      const float* ip = in + (size_t)b * CI * IH * IW + iy * IW + ix;
      const float* wp = w + ((ky * 4 + kx) * CI) * CO + co;
      for (int ci = 0; ci < CI; ++ci)
        acc += ip[(size_t)ci * IH * IW] * wp[ci * CO];
    }
  }
  out[t] = acc > 0.f ? acc : 0.f;
}

// ---------------------------------------------------------------------------
// plain 3x3 SAME conv, plane-major, OIHW weights, optional relu
// ---------------------------------------------------------------------------
__global__ void k_conv3x3(const float* __restrict__ in,
                          const float* __restrict__ w,
                          const float* __restrict__ bias,
                          float* __restrict__ out, int CI, int CO, int Hh,
                          int Ww, int relu) {
  long long t = (long long)blockIdx.x * blockDim.x + threadIdx.x;
  long long total = (long long)B_N * CO * Hh * Ww;
  if (t >= total) return;
  int b = (int)(t / ((long long)CO * Hh * Ww));
  int rem = (int)(t % ((long long)CO * Hh * Ww));
  int co = rem / (Hh * Ww);
  int p = rem % (Hh * Ww);
  int oy = p / Ww, ox = p % Ww;
  float acc = bias[co];
  for (int dy = 0; dy < 3; ++dy) {
    int iy = oy + dy - 1;
    if (iy < 0 || iy >= Hh) continue;
    for (int dx = 0; dx < 3; ++dx) {
      int ix = ox + dx - 1;
      if (ix < 0 || ix >= Ww) continue;
      for (int ci = 0; ci < CI; ++ci)
        acc += in[((size_t)b * CI + ci) * Hh * Ww + iy * Ww + ix] *
               w[((co * CI + ci) * 3 + dy) * 3 + dx];
    }
  }
  if (relu) acc = acc > 0.f ? acc : 0.f;
  out[t] = acc;
}

// ---------------------------------------------------------------------------
extern "C" void kernel_launch(void* const* d_in, const int* in_sizes, int n_in,
                              void* d_out, int out_size, void* d_ws,
                              size_t ws_size, hipStream_t stream) {
  const float* x       = (const float*)d_in[0];
  const float* ce_w1   = (const float*)d_in[1];
  const float* ce_b1   = (const float*)d_in[2];
  const float* ce_w2   = (const float*)d_in[3];
  const float* ce_b2   = (const float*)d_in[4];
  const float* ce_fcw  = (const float*)d_in[5];
  const float* ce_fcb  = (const float*)d_in[6];
  const float* ge_w1   = (const float*)d_in[7];
  const float* ge_b1   = (const float*)d_in[8];
  const float* ge_w2   = (const float*)d_in[9];
  const float* ge_b2   = (const float*)d_in[10];
  const float* ge_fcw  = (const float*)d_in[11];
  const float* ge_fcb  = (const float*)d_in[12];
  const float* mem     = (const float*)d_in[13];
  const float* dec_fcw = (const float*)d_in[14];
  const float* dec_fcb = (const float*)d_in[15];
  const float* d_w1    = (const float*)d_in[16];
  const float* d_b1    = (const float*)d_in[17];
  const float* d_w2    = (const float*)d_in[18];
  const float* d_b2    = (const float*)d_in[19];
  const float* d_w3    = (const float*)d_in[20];
  const float* d_b3    = (const float*)d_in[21];
  const float* d_w4    = (const float*)d_in[22];
  const float* d_b4    = (const float*)d_in[23];

  char* ws = (char*)d_ws;
  size_t off = 0;
  auto alloc = [&](size_t bytes) {
    size_t o = off;
    off += (bytes + 255) & ~(size_t)255;
    return o;
  };
  const size_t SZ_H1C = (size_t)B_N * 196 * 32 * 2;   // 25.7 MB
  const size_t SZ_H1G = (size_t)B_N * 784 * 32 * 2;   // 102.8 MB
  size_t o_h1c  = alloc(SZ_H1C);
  size_t o_h1g  = alloc(SZ_H1G);
  size_t o_z    = alloc((size_t)B_N * 128 * 4);
  size_t o_zn   = alloc((size_t)B_N * 128 * 2);
  size_t o_mn   = alloc((size_t)8192 * 128 * 2);
  size_t o_sim  = alloc((size_t)B_N * 8192 * 4);      // 67 MB
  size_t o_zm   = alloc((size_t)B_N * 128 * 4);
  size_t o_zm16 = alloc((size_t)B_N * 128 * 2);
  size_t o_fcw  = alloc((size_t)3136 * 128 * 2);
  size_t o_w2c  = alloc((size_t)9 * 64 * 32 * 2);
  size_t o_w2g  = alloc((size_t)9 * 64 * 32 * 2);

  _Float16* h1c   = (_Float16*)(ws + o_h1c);
  _Float16* h1g   = (_Float16*)(ws + o_h1g);
  float*    z     = (float*)(ws + o_z);
  _Float16* zn16  = (_Float16*)(ws + o_zn);
  _Float16* mn16  = (_Float16*)(ws + o_mn);
  float*    sim   = (float*)(ws + o_sim);
  float*    zm    = (float*)(ws + o_zm);
  _Float16* zm16  = (_Float16*)(ws + o_zm16);
  _Float16* fcw16 = (_Float16*)(ws + o_fcw);
  _Float16* w2c16 = (_Float16*)(ws + o_w2c);
  _Float16* w2g16 = (_Float16*)(ws + o_w2g);
  // aliases (lifetimes disjoint):
  float* d0 = (float*)(ws + o_h1c);                        // B*3136 f32
  float* t1 = (float*)(ws + o_sim);                        // B*32*196 f32
  float* t3 = (float*)(ws + o_h1g);                        // B*8*784 f32
  float* t2 = (float*)(ws + o_h1g + (size_t)B_N * 8 * 784 * 4);  // B*16*196

  // 1) weight conversions
  k_pack_w2<<<(9 * 64 * 32 + 255) / 256, 256, 0, stream>>>(ce_w2, w2c16);
  k_pack_w2<<<(9 * 64 * 32 + 255) / 256, 256, 0, stream>>>(ge_w2, w2g16);
  k_f32_to_f16<<<(3136 * 128 + 255) / 256, 256, 0, stream>>>(dec_fcw, fcw16,
                                                             3136 * 128);
  // 2) conv1 (center crop 14x14, global 28x28)
  k_conv1<<<(B_N * 196 + 255) / 256, 256, 0, stream>>>(x, 7, 7, 14, 14, ce_w1,
                                                       ce_b1, h1c);
  k_conv1<<<(B_N * 784 + 255) / 256, 256, 0, stream>>>(x, 0, 0, 28, 28, ge_w1,
                                                       ge_b1, h1g);
  // 3) fused conv2+pool+fc (WMMA implicit GEMM, async-LDS weight staging)
  k_conv2_pool_fc<<<B_N, 128, 0, stream>>>(h1c, 14, 14, w2c16, ce_b2, ce_fcw,
                                           ce_fcb, z, 0);
  k_conv2_pool_fc<<<B_N, 128, 0, stream>>>(h1g, 28, 28, w2g16, ge_b2, ge_fcw,
                                           ge_fcb, z, 64);
  // 4) normalize
  k_rownorm<<<8192, 128, 0, stream>>>(mem, mn16);
  k_rownorm<<<B_N, 128, 0, stream>>>(z, zn16);
  // 5) sim = zn @ mn^T  (M=2048, N=8192, K=128)  WMMA, 16x64 tiles
  {
    int waves = (B_N / 16) * (8192 / 64);
    k_gemm_nt<<<(waves + 7) / 8, 256, 0, stream>>>(zn16, mn16, nullptr, sim,
                                                   B_N, 8192);
  }
  // 6) top-10 + softmax + memory blend
  k_topk_match<<<B_N, 32, 0, stream>>>(sim, mem, zm, zm16);
  // 7) decoder FC (M=2048, N=3136=49*64, K=128) WMMA, 16x64 tiles
  {
    int waves = (B_N / 16) * (3136 / 64);
    k_gemm_nt<<<(waves + 7) / 8, 256, 0, stream>>>(zm16, fcw16, dec_fcb, d0,
                                                   B_N, 3136);
  }
  // 8) decoder tail
  k_deconv<<<(int)(((long long)B_N * 32 * 196 + 255) / 256), 256, 0, stream>>>(
      d0, d_w1, d_b1, t1, 64, 32, 7, 7);
  k_conv3x3<<<(int)(((long long)B_N * 16 * 196 + 255) / 256), 256, 0,
              stream>>>(t1, d_w2, d_b2, t2, 32, 16, 14, 14, 1);
  k_deconv<<<(int)(((long long)B_N * 8 * 784 + 255) / 256), 256, 0, stream>>>(
      t2, d_w3, d_b3, t3, 16, 8, 14, 14);
  k_conv3x3<<<(int)(((long long)B_N * 784 + 255) / 256), 256, 0, stream>>>(
      t3, d_w4, d_b4, (float*)d_out, 8, 1, 28, 28, 0);
}
